// DimNetPPInteraction_48490180772449
// MI455X (gfx1250) — compile-verified, matrix operand order checked by reading, not run on hardware
//
#include <hip/hip_runtime.h>
#include <hip/hip_bf16.h>
#include <stdint.h>

// ---------------------------------------------------------------------------
// DimeNet++ interaction block for MI455X (gfx1250, wave32, WMMA).
//
// - All edge-chain intermediates stored bf16; WMMA A/B fragments load as b128.
// - Triplet gather operand xd is bf16 (halves the dominant gather bandwidth).
// - Aggregation: f32 hardware atomics into L2-resident [E,64] buffer.
// - Triplet-stage sbf block staged into LDS via GLOBAL_LOAD_ASYNC_TO_LDS_B128
//   (ASYNCcnt path), waited with s_wait_asynccnt.
// ---------------------------------------------------------------------------

typedef __attribute__((ext_vector_type(16))) __bf16 v16bf;
typedef __attribute__((ext_vector_type(8)))  float  v8f;

union V16 { uint4 q[2]; unsigned short s[16]; v16bf v; };

__device__ __forceinline__ unsigned short f2bf(float f) {
    uint32_t u = __float_as_uint(f);
    return (unsigned short)((u + 0x7FFFu + ((u >> 16) & 1u)) >> 16);  // RNE
}
__device__ __forceinline__ float bf2f(unsigned short s) {
    return __uint_as_float(((uint32_t)s) << 16);
}
__device__ __forceinline__ float silu_f(float x) {
    return x * (1.0f / (1.0f + __expf(-x)));
}

// ---------------------------------------------------------------------------
// fp32 [n] -> bf16 [n], 4 elements per thread (n must be a multiple of 4)
// ---------------------------------------------------------------------------
__global__ __launch_bounds__(256) void cvt_f32_bf16_kernel(
    const float* __restrict__ src, unsigned short* __restrict__ dst, long n4) {
    long i = (long)blockIdx.x * 256 + threadIdx.x;
    if (i >= n4) return;
    float4 v = ((const float4*)src)[i];
    uint2 p;
    p.x = (uint32_t)f2bf(v.x) | ((uint32_t)f2bf(v.y) << 16);
    p.y = (uint32_t)f2bf(v.z) | ((uint32_t)f2bf(v.w) << 16);
    ((uint2*)dst)[i] = p;
}

// ---------------------------------------------------------------------------
// Weight prep: src fp32 row-major [K][N] -> dst bf16 [N][K] (transposed),
// so WMMA B-fragments read 16 contiguous bf16 per lane.
// ---------------------------------------------------------------------------
__global__ __launch_bounds__(256) void wprep_kernel(const float* __restrict__ src,
                                                    unsigned short* __restrict__ dst,
                                                    int K, int N) {
    int i = blockIdx.x * 256 + threadIdx.x;
    if (i >= K * N) return;
    int k = i / N, n = i % N;
    dst[(size_t)n * K + k] = f2bf(src[i]);
}

// ---------------------------------------------------------------------------
// Fused GEMM: out[r,:] = (res ? res[r,:] : 0) + silu(A[r,:] @ W + bias)
//   A   : bf16 [nrows,K]  (A_BF16=true)  or fp32 [nrows,K] (A_BF16=false)
//   Wt  : bf16 [N,K] transposed
//   res : bf16 or null
//   out : f32 (OUT_F32=true, final stage) or bf16
// Block = 256 threads = 8 waves; tile = 128 rows x N cols; wave w owns rows
// [blk*128+16w, +16) and all N/16 WMMA tiles; K-loop in chunks of 32.
// ---------------------------------------------------------------------------
template <int K, int N, bool A_BF16, bool OUT_F32>
__global__ __launch_bounds__(256) void gemm_bias_silu_res(
    const void* __restrict__ A_, const unsigned short* __restrict__ Wt,
    const float* __restrict__ bias, const unsigned short* __restrict__ res,
    void* __restrict__ out_, int nrows) {

    constexpr int NT = N / 16;
    const int wave  = threadIdx.x >> 5;
    const int lane  = threadIdx.x & 31;
    const int mlane = lane & 15;
    const int khalf = (lane >> 4) << 4;   // 0 or 16: K-half held by this lane
    const int r0    = blockIdx.x * 128 + wave * 16;

    v8f acc[NT];
#pragma unroll
    for (int nt = 0; nt < NT; ++nt) {
        v8f z = {0.f, 0.f, 0.f, 0.f, 0.f, 0.f, 0.f, 0.f};
        acc[nt] = z;
    }

    const int  arow   = r0 + mlane;
    const bool avalid = (arow < nrows);

#pragma unroll
    for (int kc = 0; kc < K; kc += 32) {
        const int kb = kc + khalf;

        // ---- A fragment (16 contiguous K-values of row `arow`)
        V16 afrag;
        if (A_BF16) {
            if (avalid) {
                const uint4* ap =
                    (const uint4*)((const unsigned short*)A_ + (size_t)arow * K + kb);
                afrag.q[0] = ap[0];
                afrag.q[1] = ap[1];
            } else {
                uint4 z = {0u, 0u, 0u, 0u};
                afrag.q[0] = z;
                afrag.q[1] = z;
            }
        } else {
            float av[16];
            if (avalid) {
                const float4* ap = (const float4*)((const float*)A_ + (size_t)arow * K + kb);
                float4 t0 = ap[0], t1 = ap[1], t2 = ap[2], t3 = ap[3];
                av[0]=t0.x; av[1]=t0.y; av[2]=t0.z; av[3]=t0.w;
                av[4]=t1.x; av[5]=t1.y; av[6]=t1.z; av[7]=t1.w;
                av[8]=t2.x; av[9]=t2.y; av[10]=t2.z; av[11]=t2.w;
                av[12]=t3.x; av[13]=t3.y; av[14]=t3.z; av[15]=t3.w;
            } else {
#pragma unroll
                for (int i = 0; i < 16; ++i) av[i] = 0.f;
            }
#pragma unroll
            for (int i = 0; i < 16; ++i) afrag.s[i] = f2bf(av[i]);
        }

        // ---- N/16 WMMA tiles; B frag = 16 contiguous bf16 of transposed W
#pragma unroll
        for (int nt = 0; nt < NT; ++nt) {
            const uint4* wp =
                (const uint4*)(Wt + ((size_t)(nt * 16 + mlane)) * K + kb);
            V16 bfrag;
            bfrag.q[0] = wp[0];
            bfrag.q[1] = wp[1];
            acc[nt] = __builtin_amdgcn_wmma_f32_16x16x32_bf16(
                false, afrag.v, false, bfrag.v, (short)0, acc[nt], false, false);
        }
    }

    // ---- epilogue: bias + silu + optional bf16 residual, store f32/bf16
#pragma unroll
    for (int nt = 0; nt < NT; ++nt) {
        const int col = nt * 16 + mlane;
        const float bv = bias ? bias[col] : 0.f;
#pragma unroll
        for (int v = 0; v < 8; ++v) {
            const int row = r0 + v + ((lane >> 4) << 3);
            if (row < nrows) {
                float a = silu_f(acc[nt][v] + bv);
                if (res) a += bf2f(res[(size_t)row * N + col]);
                if (OUT_F32)
                    ((float*)out_)[(size_t)row * N + col] = a;
                else
                    ((unsigned short*)out_)[(size_t)row * N + col] = f2bf(a);
            }
        }
    }
}

// ---------------------------------------------------------------------------
// rbf basis path (fused, elementwise): buf[e,c] *= ((rbf[e]@w1)@w2)[c]
// buf is bf16; 2 edges per block; thread handles one of 128 columns.
// ---------------------------------------------------------------------------
__global__ __launch_bounds__(256) void rbf_mul_kernel(
    const float* __restrict__ rbf, const float* __restrict__ w1,
    const float* __restrict__ w2, unsigned short* __restrict__ buf, int nE) {
    const int e = blockIdx.x * 2 + (threadIdx.x >> 7);
    const int c = threadIdx.x & 127;
    if (e >= nE) return;
    float r8[8] = {0.f, 0.f, 0.f, 0.f, 0.f, 0.f, 0.f, 0.f};
#pragma unroll
    for (int i = 0; i < 6; ++i) {
        const float v = rbf[(size_t)e * 6 + i];
#pragma unroll
        for (int k = 0; k < 8; ++k) r8[k] += v * w1[i * 8 + k];
    }
    float s = 0.f;
#pragma unroll
    for (int k = 0; k < 8; ++k) s += r8[k] * w2[k * 128 + c];
    const size_t o = (size_t)e * 128 + c;
    buf[o] = f2bf(bf2f(buf[o]) * s);
}

// ---------------------------------------------------------------------------
// Triplet stage: agg[idx_ji[t]] += xd[idx_kj[t]] * ((sbf[t]@w1)@w2)
// 64 triplets/block. sbf block (64x42 fp32 = 10.5 KB) staged into LDS with
// GLOBAL_LOAD_ASYNC_TO_LDS_B128 (ASYNCcnt, no VGPR round-trip); basis weights
// staged with normal loads; s_wait_asynccnt 0 + barrier before compute.
// 4 lanes per triplet, 16 of 64 channels each; xd gathered as bf16; f32
// hardware atomics into L2-resident agg.
// ---------------------------------------------------------------------------
__global__ __launch_bounds__(256) void triplet_kernel(
    const float* __restrict__ sbf, const int* __restrict__ idx_kj,
    const int* __restrict__ idx_ji, const float* __restrict__ w1,
    const float* __restrict__ w2, const unsigned short* __restrict__ xd,
    float* __restrict__ agg, int nT) {

    __shared__ float s_sbf[64 * 42];
    __shared__ float s_w1[42 * 8];
    __shared__ float s_w2[8 * 64];

    const int tid = threadIdx.x;
    const long base = (long)blockIdx.x * 64;
    const long gbase = base * 42;
    const long glim  = (long)nT * 42;
    const int  nflt  = (int)((glim - gbase) < (long)(64 * 42) ? (glim - gbase)
                                                              : (long)(64 * 42));

    // ---- async stage: 64*42 fp32 = 672 b128 transfers, 256 lanes
    const float* __restrict__ gsrc = sbf + gbase;
    for (int i = tid; i < (64 * 42) / 4; i += 256) {
        const int f = i * 4;
        if (f + 4 <= nflt) {
            const unsigned lds_off = (unsigned)(uintptr_t)(&s_sbf[f]);
            const float* gp = gsrc + f;
            asm volatile("global_load_async_to_lds_b128 %0, %1, off"
                         :: "v"(lds_off), "v"(gp)
                         : "memory");
        } else {  // partial tail block (nT not a multiple of 64)
            for (int j = 0; j < 4; ++j)
                s_sbf[f + j] = (f + j < nflt) ? gsrc[f + j] : 0.f;
        }
    }

    // ---- stage basis weights (small, normal path)
    for (int i = tid; i < 42 * 8; i += 256) s_w1[i] = w1[i];
    for (int i = tid; i < 8 * 64; i += 256) s_w2[i] = w2[i];

    asm volatile("s_wait_asynccnt 0" ::: "memory");
    __syncthreads();

    const int slot = tid >> 2;
    const int q    = tid & 3;
    const long t   = base + slot;
    if (t >= nT) return;

    const int kj = idx_kj[t];
    const int ji = idx_ji[t];
    const unsigned short* __restrict__ xrow = xd + (size_t)kj * 64 + q * 16;
    __builtin_prefetch(xrow, 0, 0);  // global_prefetch_b8: warm the gather row

    float t8[8] = {0.f, 0.f, 0.f, 0.f, 0.f, 0.f, 0.f, 0.f};
#pragma unroll
    for (int i = 0; i < 42; ++i) {
        const float s = s_sbf[slot * 42 + i];
#pragma unroll
        for (int k = 0; k < 8; ++k) t8[k] += s * s_w1[i * 8 + k];
    }

    // gather 16 bf16 channels (32 bytes) and widen
    V16 xv;
    const uint4* xp = (const uint4*)xrow;
    xv.q[0] = xp[0];
    xv.q[1] = xp[1];

    float* __restrict__ arow = agg + (size_t)ji * 64 + q * 16;
#pragma unroll
    for (int jj = 0; jj < 16; ++jj) {
        const int j = q * 16 + jj;
        float se = 0.f;
#pragma unroll
        for (int k = 0; k < 8; ++k) se += t8[k] * s_w2[k * 64 + j];
        unsafeAtomicAdd(arow + jj, bf2f(xv.s[jj]) * se);  // global_atomic_add_f32
    }
}

// ---------------------------------------------------------------------------
// Host launch
// ---------------------------------------------------------------------------
extern "C" void kernel_launch(void* const* d_in, const int* in_sizes, int n_in,
                              void* d_out, int out_size, void* d_ws, size_t ws_size,
                              hipStream_t stream) {
    const int D = 128, DD = 64;
    const int nE = in_sizes[0] / D;   // 300000
    const int nT = in_sizes[3];       // 3000000

    const float* x      = (const float*)d_in[0];
    const float* rbf    = (const float*)d_in[1];
    const float* sbf    = (const float*)d_in[2];
    const int*   idx_kj = (const int*)d_in[3];
    const int*   idx_ji = (const int*)d_in[4];
    const float* w_rbf1 = (const float*)d_in[5];
    const float* w_rbf2 = (const float*)d_in[6];
    const float* w_sbf1 = (const float*)d_in[7];
    const float* w_sbf2 = (const float*)d_in[8];
    const float* w_kj   = (const float*)d_in[9];
    const float* b_kj   = (const float*)d_in[10];
    const float* w_ji   = (const float*)d_in[11];
    const float* b_ji   = (const float*)d_in[12];
    const float* w_down = (const float*)d_in[13];
    const float* w_up   = (const float*)d_in[14];
    const float* r1w1 = (const float*)d_in[15]; const float* r1b1 = (const float*)d_in[16];
    const float* r1w2 = (const float*)d_in[17]; const float* r1b2 = (const float*)d_in[18];
    const float* w_bs = (const float*)d_in[19]; const float* b_bs = (const float*)d_in[20];
    const float* r2w1 = (const float*)d_in[21]; const float* r2b1 = (const float*)d_in[22];
    const float* r2w2 = (const float*)d_in[23]; const float* r2b2 = (const float*)d_in[24];
    const float* r3w1 = (const float*)d_in[25]; const float* r3b1 = (const float*)d_in[26];
    const float* r3w2 = (const float*)d_in[27]; const float* r3b2 = (const float*)d_in[28];

    // ---- workspace carve-out
    size_t off = 0;
    auto carve = [&](size_t bytes) -> void* {
        void* p = (char*)d_ws + off;
        off += (bytes + 255) & ~(size_t)255;
        return p;
    };
    unsigned short* xbf  = (unsigned short*)carve((size_t)nE * D * 2);
    unsigned short* bufA = (unsigned short*)carve((size_t)nE * D * 2);
    unsigned short* bufB = (unsigned short*)carve((size_t)nE * D * 2);
    unsigned short* bufC = (unsigned short*)carve((size_t)nE * D * 2);
    unsigned short* xd   = (unsigned short*)carve((size_t)nE * DD * 2);
    float*          agg  = (float*)carve((size_t)nE * DD * 4);
    unsigned short* wt_ji   = (unsigned short*)carve((size_t)D * D * 2);
    unsigned short* wt_kj   = (unsigned short*)carve((size_t)D * D * 2);
    unsigned short* wt_down = (unsigned short*)carve((size_t)D * DD * 2);
    unsigned short* wt_up   = (unsigned short*)carve((size_t)DD * D * 2);
    unsigned short* wt_r1a  = (unsigned short*)carve((size_t)D * D * 2);
    unsigned short* wt_r1b  = (unsigned short*)carve((size_t)D * D * 2);
    unsigned short* wt_bs   = (unsigned short*)carve((size_t)D * D * 2);
    unsigned short* wt_r2a  = (unsigned short*)carve((size_t)D * D * 2);
    unsigned short* wt_r2b  = (unsigned short*)carve((size_t)D * D * 2);
    unsigned short* wt_r3a  = (unsigned short*)carve((size_t)D * D * 2);
    unsigned short* wt_r3b  = (unsigned short*)carve((size_t)D * D * 2);

    // ---- weight prep (bf16 transposed) + x -> bf16
    auto prep = [&](const float* s, unsigned short* dst, int K, int N) {
        int n = K * N;
        wprep_kernel<<<(n + 255) / 256, 256, 0, stream>>>(s, dst, K, N);
    };
    prep(w_ji, wt_ji, D, D);
    prep(w_kj, wt_kj, D, D);
    prep(w_down, wt_down, D, DD);
    prep(w_up, wt_up, DD, D);
    prep(r1w1, wt_r1a, D, D);
    prep(r1w2, wt_r1b, D, D);
    prep(w_bs, wt_bs, D, D);
    prep(r2w1, wt_r2a, D, D);
    prep(r2w2, wt_r2b, D, D);
    prep(r3w1, wt_r3a, D, D);
    prep(r3w2, wt_r3b, D, D);

    {
        long n4 = (long)nE * D / 4;
        cvt_f32_bf16_kernel<<<(int)((n4 + 255) / 256), 256, 0, stream>>>(x, xbf, n4);
    }

    hipMemsetAsync(agg, 0, (size_t)nE * DD * 4, stream);

    const int GB = (nE + 127) / 128;  // GEMM grid (128-row tiles)

    // 1. t_ji = silu(x @ w_ji + b_ji)              -> bufA (bf16)
    gemm_bias_silu_res<128, 128, true, false>
        <<<GB, 256, 0, stream>>>(xbf, wt_ji, b_ji, nullptr, bufA, nE);
    // 2. t_kj = silu(x @ w_kj + b_kj)              -> bufB (bf16)
    gemm_bias_silu_res<128, 128, true, false>
        <<<GB, 256, 0, stream>>>(xbf, wt_kj, b_kj, nullptr, bufB, nE);
    // 3. bufB *= rbf_e (in-place, bf16)
    rbf_mul_kernel<<<(nE + 1) / 2, 256, 0, stream>>>(rbf, w_rbf1, w_rbf2, bufB, nE);
    // 4. xd = silu(bufB @ w_down)                  -> xd (bf16, [E,64])
    gemm_bias_silu_res<128, 64, true, false>
        <<<GB, 256, 0, stream>>>(bufB, wt_down, nullptr, nullptr, xd, nE);
    // 5. triplet gather/scatter                    -> agg (f32)
    triplet_kernel<<<(nT + 63) / 64, 256, 0, stream>>>(sbf, idx_kj, idx_ji,
                                                       w_sbf1, w_sbf2, xd, agg, nT);
    // 6. h0 = silu(agg @ w_up) + t_ji              -> bufB
    gemm_bias_silu_res<64, 128, false, false>
        <<<GB, 256, 0, stream>>>(agg, wt_up, nullptr, bufA, bufB, nE);
    // 7. res1: t = silu(h0 @ w1 + b1)              -> bufA
    gemm_bias_silu_res<128, 128, true, false>
        <<<GB, 256, 0, stream>>>(bufB, wt_r1a, r1b1, nullptr, bufA, nE);
    // 8.       h1 = h0 + silu(t @ w2 + b2)         -> bufC
    gemm_bias_silu_res<128, 128, true, false>
        <<<GB, 256, 0, stream>>>(bufA, wt_r1b, r1b2, bufB, bufC, nE);
    // 9. h2 = silu(h1 @ w_bs + b_bs) + x           -> bufA
    gemm_bias_silu_res<128, 128, true, false>
        <<<GB, 256, 0, stream>>>(bufC, wt_bs, b_bs, xbf, bufA, nE);
    // 10. res2: t = silu(h2 @ w1 + b1)             -> bufB
    gemm_bias_silu_res<128, 128, true, false>
        <<<GB, 256, 0, stream>>>(bufA, wt_r2a, r2b1, nullptr, bufB, nE);
    // 11.       h3 = h2 + silu(t @ w2 + b2)        -> bufC
    gemm_bias_silu_res<128, 128, true, false>
        <<<GB, 256, 0, stream>>>(bufB, wt_r2b, r2b2, bufA, bufC, nE);
    // 12. res3: t = silu(h3 @ w1 + b1)             -> bufA
    gemm_bias_silu_res<128, 128, true, false>
        <<<GB, 256, 0, stream>>>(bufC, wt_r3a, r3b1, nullptr, bufA, nE);
    // 13.       out = h3 + silu(t @ w2 + b2)       -> d_out (f32)
    gemm_bias_silu_res<128, 128, true, true>
        <<<GB, 256, 0, stream>>>(bufA, wt_r3b, r3b2, bufC, d_out, nE);
}
